// DynamicBlock_87436944212454
// MI455X (gfx1250) — matrix operand order repaired and checked by reading, not compiled, not run on hardware
//
#include <hip/hip_runtime.h>

// ---------------- problem constants ----------------
#define BATCH 4
#define SEQ   2048
#define DIM   2048
#define NH    16
#define NKV   4
#define HDIM  128
#define ISZ   5632
#define NTOK  2048
#define QCOLS (NH * HDIM)    // 2048
#define KCOLS (NKV * HDIM)   // 512
#define RMS_EPS 1e-6f

// ---------------- WMMA types ----------------
typedef __attribute__((ext_vector_type(16))) __bf16 v16bf;
typedef __attribute__((ext_vector_type(8)))  float  v8f;
typedef __attribute__((ext_vector_type(4)))  unsigned int v4u;
typedef __attribute__((ext_vector_type(4)))  int v4i;

union BF16Frag { v16bf v; v4u u[2]; };

__device__ __forceinline__ unsigned short f2bf(float f) {
    unsigned int u = __float_as_uint(f);
    u += 0x7FFFu + ((u >> 16) & 1u);      // round to nearest even
    return (unsigned short)(u >> 16);
}
__device__ __forceinline__ float bf2f(unsigned short h) {
    return __uint_as_float(((unsigned int)h) << 16);
}
__device__ __forceinline__ unsigned pack2(float lo, float hi) {
    return (unsigned)f2bf(lo) | ((unsigned)f2bf(hi) << 16);
}

// ---------------- CDNA5 async global->LDS copy (guarded) ----------------
#if defined(__gfx1250__) && __has_builtin(__builtin_amdgcn_global_load_async_to_lds_b128)
#define HAVE_ASYNC_LDS 1
#else
#define HAVE_ASYNC_LDS 0
#endif

#if HAVE_ASYNC_LDS
__device__ __forceinline__ void async_copy16(const void* g, void* l) {
    // prototype: (v4i addrspace(1)*, v4i addrspace(3)*, imm offset, imm cpol)
    typedef __attribute__((address_space(1))) v4i* gp_t;
    typedef __attribute__((address_space(3))) v4i* lp_t;
    // generic->AS pointer via integer round-trip (low 32 bits of a generic
    // LDS address are the LDS byte offset per CDNA5 aperture rules)
    __builtin_amdgcn_global_load_async_to_lds_b128(
        (gp_t)(unsigned long long)g,
        (lp_t)(unsigned int)(unsigned long long)l, 0, 0);
}
__device__ __forceinline__ void async_wait0() {
#if __has_builtin(__builtin_amdgcn_s_wait_asynccnt)
    __builtin_amdgcn_s_wait_asynccnt(0);
#else
    asm volatile("s_wait_asynccnt 0x0" ::: "memory");
#endif
}
#endif

// =====================================================================
// Tiled GEMM:  C[M,N] = A[M,K](bf16) * B[K,N](fp32 weights, converted
// to bf16 while staging to LDS) + optional bias/residual, with fused
// epilogues. 128x128 tile, K-step 32, 8 waves (wave32) as 4x2 grid.
// Double-buffered LDS, one barrier per K-tile: global fetches for tile
// kt+1 are issued before the WMMAs of tile kt so HBM/L2 latency hides
// behind matrix math. A tiles go via GLOBAL_LOAD_ASYNC_TO_LDS_B128 when
// the toolchain exposes it.
// EPI: 0 = store fp32 (bias/resid optional)
//      2 = store bf16( silu(acc) )
//      3 = store bf16( acc * bf2f(gact) )
// =====================================================================
#define LDA 48   // ushort stride (32 K + 16 pad), 96B = 6*16B aligned
#define LDB 48

__device__ __forceinline__ void store_b_packed(unsigned short* bs, int cseg, int k2,
                                               float4 q0, float4 q1, float4 q2, float4 q3) {
    // bs layout [n][k]; pack rows (k2, k2+1) of one column into one b32
    *(unsigned*)&bs[(cseg + 0) * LDB + k2] = pack2(q0.x, q2.x);
    *(unsigned*)&bs[(cseg + 1) * LDB + k2] = pack2(q0.y, q2.y);
    *(unsigned*)&bs[(cseg + 2) * LDB + k2] = pack2(q0.z, q2.z);
    *(unsigned*)&bs[(cseg + 3) * LDB + k2] = pack2(q0.w, q2.w);
    *(unsigned*)&bs[(cseg + 4) * LDB + k2] = pack2(q1.x, q3.x);
    *(unsigned*)&bs[(cseg + 5) * LDB + k2] = pack2(q1.y, q3.y);
    *(unsigned*)&bs[(cseg + 6) * LDB + k2] = pack2(q1.z, q3.z);
    *(unsigned*)&bs[(cseg + 7) * LDB + k2] = pack2(q1.w, q3.w);
}

template<int EPI>
__global__ __launch_bounds__(256) void gemm_bf16_kernel(
    const unsigned short* __restrict__ A,   // [M,K] bf16
    const float*          __restrict__ B,   // [K,N] fp32
    const float*          __restrict__ bias,// [N] or null
    const float*          __restrict__ resid,// [M,N] or null
    const unsigned short* __restrict__ gact,// [M,N] bf16 (EPI==3)
    float*                __restrict__ Cf,  // fp32 out (EPI==0)
    unsigned short*       __restrict__ Cb,  // bf16 out (EPI==2,3)
    int M, int Nn, int K)
{
    __shared__ unsigned short As[2][128 * LDA];
    __shared__ unsigned short Bs[2][128 * LDB];

    const int tid  = threadIdx.x;
    const int wid  = tid >> 5;
    const int lane = tid & 31;
    const int l16  = lane & 15;
    const int half = lane >> 4;

    const int m0 = blockIdx.y * 128;
    const int n0 = blockIdx.x * 128;
    const int wm = (wid & 3) * 32;   // wave row offset inside tile
    const int wn = (wid >> 2) * 64;  // wave col offset inside tile

    v8f acc[2][4];
#pragma unroll
    for (int i = 0; i < 2; ++i)
#pragma unroll
        for (int j = 0; j < 4; ++j)
#pragma unroll
            for (int e = 0; e < 8; ++e) acc[i][j][e] = 0.0f;

    // A staging: thread owns 16 contiguous bf16 of one row
    const int ar = tid >> 1;
    const int ac = (tid & 1) * 16;
    // B staging: thread owns K-row pair (k2, k2+1) x 8 columns
    const int kp   = tid >> 4;
    const int cseg = (tid & 15) * 8;
    const int k2   = kp * 2;

    const unsigned short* Abase = A + (size_t)(m0 + ar) * K + ac;
    const float*          Bbase = B + (size_t)k2 * Nn + n0 + cseg;

    const int nk = K >> 5;

    // ---- prologue: stage tile 0 into buffer 0 ----
    {
#if HAVE_ASYNC_LDS
        async_copy16(Abase,     &As[0][ar * LDA + ac]);
        async_copy16(Abase + 8, &As[0][ar * LDA + ac + 8]);
#else
        v4u a0 = *(const v4u*)Abase;
        v4u a1 = *(const v4u*)(Abase + 8);
        *(v4u*)&As[0][ar * LDA + ac]     = a0;
        *(v4u*)&As[0][ar * LDA + ac + 8] = a1;
#endif
        float4 q0 = *(const float4*)(Bbase);
        float4 q1 = *(const float4*)(Bbase + 4);
        float4 q2 = *(const float4*)(Bbase + Nn);
        float4 q3 = *(const float4*)(Bbase + Nn + 4);
        store_b_packed(Bs[0], cseg, k2, q0, q1, q2, q3);
#if HAVE_ASYNC_LDS
        async_wait0();
#endif
        __syncthreads();
    }

    for (int kt = 0; kt < nk; ++kt) {
        const int  b    = kt & 1;
        const bool more = (kt + 1 < nk);

#if !HAVE_ASYNC_LDS
        v4u a0 = {}, a1 = {};
#endif
        float4 q0 = {}, q1 = {}, q2 = {}, q3 = {};
        if (more) {
            const unsigned short* ag = Abase + (size_t)(kt + 1) * 32;
#if HAVE_ASYNC_LDS
            async_copy16(ag,     &As[b ^ 1][ar * LDA + ac]);
            async_copy16(ag + 8, &As[b ^ 1][ar * LDA + ac + 8]);
#else
            a0 = *(const v4u*)ag;
            a1 = *(const v4u*)(ag + 8);
#endif
            const float* bg = Bbase + (size_t)(kt + 1) * 32 * Nn;
            q0 = *(const float4*)(bg);
            q1 = *(const float4*)(bg + 4);
            q2 = *(const float4*)(bg + Nn);
            q3 = *(const float4*)(bg + Nn + 4);
        }

        // ---- compute tile kt from buffer b ----
        BF16Frag af[2], bf[4];
#pragma unroll
        for (int i = 0; i < 2; ++i) {
            const int r = wm + i * 16 + l16;
            af[i].u[0] = *(const v4u*)&As[b][r * LDA + half * 8];
            af[i].u[1] = *(const v4u*)&As[b][r * LDA + half * 8 + 16];
        }
#pragma unroll
        for (int j = 0; j < 4; ++j) {
            const int cc = wn + j * 16 + l16;
            bf[j].u[0] = *(const v4u*)&Bs[b][cc * LDB + half * 8];
            bf[j].u[1] = *(const v4u*)&Bs[b][cc * LDB + half * 8 + 16];
        }
#pragma unroll
        for (int i = 0; i < 2; ++i)
#pragma unroll
            for (int j = 0; j < 4; ++j)
                acc[i][j] = __builtin_amdgcn_wmma_f32_16x16x32_bf16(
                    false, af[i].v, false, bf[j].v, (short)0, acc[i][j], false, false);

        // ---- publish tile kt+1 into buffer b^1 ----
        if (more) {
#if !HAVE_ASYNC_LDS
            *(v4u*)&As[b ^ 1][ar * LDA + ac]     = a0;
            *(v4u*)&As[b ^ 1][ar * LDA + ac + 8] = a1;
#endif
            store_b_packed(Bs[b ^ 1], cseg, k2, q0, q1, q2, q3);
#if HAVE_ASYNC_LDS
            async_wait0();
#endif
        }
        __syncthreads();
    }

    // epilogue (C layout: VGPR r -> row r + half*8; lane l16 -> col)
#pragma unroll
    for (int i = 0; i < 2; ++i)
#pragma unroll
        for (int j = 0; j < 4; ++j) {
            const int n = n0 + wn + j * 16 + l16;
            const float bn = bias ? bias[n] : 0.0f;
#pragma unroll
            for (int r = 0; r < 8; ++r) {
                const int m = m0 + wm + i * 16 + half * 8 + r;
                const size_t idx = (size_t)m * Nn + n;
                float v = acc[i][j][r] + bn;
                if (resid) v += resid[idx];
                if (EPI == 0) {
                    Cf[idx] = v;
                } else if (EPI == 2) {
                    Cb[idx] = f2bf(v / (1.0f + __expf(-v)));   // silu
                } else {
                    Cb[idx] = f2bf(v * bf2f(gact[idx]));
                }
            }
        }
}

// =====================================================================
// Flash-style GQA attention. Block = 4 waves (128 thr) = 64 queries,
// one head per blockIdx.y. Wave owns 16 queries x HD=128 accumulator.
// =====================================================================
__global__ __launch_bounds__(128) void attn_kernel(
    const unsigned short* __restrict__ Q,   // [NTOK, QCOLS] bf16
    const unsigned short* __restrict__ Kk,  // [NTOK, KCOLS] bf16
    const unsigned short* __restrict__ Vv,  // [NTOK, KCOLS] bf16
    float*                __restrict__ O)   // [NTOK, QCOLS] fp32
{
    __shared__ unsigned short Ks[64 * 136];     // [key][hd], pad 8
    __shared__ unsigned short Vt[128 * 72];     // [hd][key], pad 8
    __shared__ unsigned short Ps[4][16 * 72];   // per-wave P tile

    const int tid  = threadIdx.x;
    const int wid  = tid >> 5;
    const int lane = tid & 31;
    const int l16  = lane & 15;
    const int half = lane >> 4;

    const int h    = blockIdx.y;
    const int kvh  = h >> 2;                    // GQA: 4 q-heads per kv-head
    const int qrow = blockIdx.x * 64 + wid * 16;

    // Q fragments: 4 K-chunks of 32 along HD, A-layout
    BF16Frag qf[4];
    {
        const unsigned short* qg = Q + (size_t)(qrow + l16) * QCOLS + h * HDIM;
#pragma unroll
        for (int c = 0; c < 4; ++c) {
            qf[c].u[0] = *(const v4u*)(qg + c * 32 + half * 8);
            qf[c].u[1] = *(const v4u*)(qg + c * 32 + half * 8 + 16);
        }
    }

    v8f oacc[8];
#pragma unroll
    for (int c = 0; c < 8; ++c)
#pragma unroll
        for (int e = 0; e < 8; ++e) oacc[c][e] = 0.0f;
    float mrow[8], lrow[8];
#pragma unroll
    for (int r = 0; r < 8; ++r) { mrow[r] = -1e30f; lrow[r] = 0.0f; }

    const float scale = 0.08838834764831845f;   // 1/sqrt(128)

    for (int kt = 0; kt < NTOK / 64; ++kt) {
        const int kbase = kt * 64;
        __syncthreads();
        {   // stage K [key][hd] (async when available) and V transposed [hd][key]
            const int row = tid >> 1;
            const int seg = (tid & 1) * 64;
            const unsigned short* kg = Kk + (size_t)(kbase + row) * KCOLS + kvh * HDIM + seg;
            const unsigned short* vg = Vv + (size_t)(kbase + row) * KCOLS + kvh * HDIM + seg;
#pragma unroll
            for (int e = 0; e < 64; e += 8) {
#if HAVE_ASYNC_LDS
                async_copy16(kg + e, &Ks[row * 136 + seg + e]);
#else
                *(v4u*)&Ks[row * 136 + seg + e] = *(const v4u*)(kg + e);
#endif
            }
#pragma unroll
            for (int e = 0; e < 64; e += 8) {
                unsigned short tmp[8];
                *(v4u*)tmp = *(const v4u*)(vg + e);
#pragma unroll
                for (int q2 = 0; q2 < 8; ++q2)
                    Vt[(seg + e + q2) * 72 + row] = tmp[q2];
            }
#if HAVE_ASYNC_LDS
            async_wait0();
#endif
        }
        __syncthreads();

        // S = Q K^T : 4 key sub-tiles x 4 HD chunks
        v8f sacc[4];
#pragma unroll
        for (int j = 0; j < 4; ++j) {
#pragma unroll
            for (int e = 0; e < 8; ++e) sacc[j][e] = 0.0f;
            const int kcol = j * 16 + l16;
#pragma unroll
            for (int c = 0; c < 4; ++c) {
                BF16Frag kf;
                kf.u[0] = *(const v4u*)&Ks[kcol * 136 + c * 32 + half * 8];
                kf.u[1] = *(const v4u*)&Ks[kcol * 136 + c * 32 + half * 8 + 16];
                sacc[j] = __builtin_amdgcn_wmma_f32_16x16x32_bf16(
                    false, qf[c].v, false, kf.v, (short)0, sacc[j], false, false);
            }
        }

        // online softmax per owned row (row = half*8 + r)
        float alpha[8];
#pragma unroll
        for (int r = 0; r < 8; ++r) {
            float mx = -1e30f;
#pragma unroll
            for (int j = 0; j < 4; ++j) mx = fmaxf(mx, sacc[j][r] * scale);
            for (int off = 1; off < 16; off <<= 1)
                mx = fmaxf(mx, __shfl_xor(mx, off, 32));
            const float mnew = fmaxf(mrow[r], mx);
            alpha[r] = __expf(mrow[r] - mnew);
            mrow[r] = mnew;
            float rs = 0.0f;
#pragma unroll
            for (int j = 0; j < 4; ++j) {
                const float p = __expf(sacc[j][r] * scale - mnew);
                sacc[j][r] = p;
                rs += p;
            }
            for (int off = 1; off < 16; off <<= 1)
                rs += __shfl_xor(rs, off, 32);
            lrow[r] = lrow[r] * alpha[r] + rs;
        }
#pragma unroll
        for (int c = 0; c < 8; ++c)
#pragma unroll
            for (int r = 0; r < 8; ++r) oacc[c][r] *= alpha[r];

        // C-layout P -> LDS -> A-layout fragments
#pragma unroll
        for (int j = 0; j < 4; ++j)
#pragma unroll
            for (int r = 0; r < 8; ++r)
                Ps[wid][(half * 8 + r) * 72 + j * 16 + l16] = f2bf(sacc[j][r]);

        // O += P V : 2 key chunks of 32 x 8 HD sub-tiles
#pragma unroll
        for (int ck = 0; ck < 2; ++ck) {
            BF16Frag pf;
            pf.u[0] = *(const v4u*)&Ps[wid][l16 * 72 + ck * 32 + half * 8];
            pf.u[1] = *(const v4u*)&Ps[wid][l16 * 72 + ck * 32 + half * 8 + 16];
#pragma unroll
            for (int c = 0; c < 8; ++c) {
                BF16Frag vf;
                const int hd = c * 16 + l16;
                vf.u[0] = *(const v4u*)&Vt[hd * 72 + ck * 32 + half * 8];
                vf.u[1] = *(const v4u*)&Vt[hd * 72 + ck * 32 + half * 8 + 16];
                oacc[c] = __builtin_amdgcn_wmma_f32_16x16x32_bf16(
                    false, pf.v, false, vf.v, (short)0, oacc[c], false, false);
            }
        }
    }

    // normalize and store
#pragma unroll
    for (int c = 0; c < 8; ++c)
#pragma unroll
        for (int r = 0; r < 8; ++r) {
            const int m   = qrow + half * 8 + r;
            const int col = h * HDIM + c * 16 + l16;
            O[(size_t)m * QCOLS + col] = oacc[c][r] / lrow[r];
        }
}

// ---------------- elementwise kernels ----------------
__global__ __launch_bounds__(256) void gather_kernel(
    const float* __restrict__ Hs, const int* __restrict__ bi,
    const int* __restrict__ ti, float* __restrict__ sel, float* __restrict__ x)
{
    const int n = blockIdx.x;
    const size_t src = ((size_t)bi[n] * SEQ + ti[n]) * DIM;
    for (int c = threadIdx.x; c < DIM; c += 256) {
        const float v = Hs[src + c];
        sel[(size_t)n * DIM + c] = v;
        x[(size_t)n * DIM + c]   = v;
    }
}

__global__ __launch_bounds__(256) void rmsnorm_kernel(
    const float* __restrict__ X, const float* __restrict__ W,
    unsigned short* __restrict__ Yb)
{
    const int row = blockIdx.x;
    const float* x = X + (size_t)row * DIM;
    float ss = 0.0f;
    for (int c = threadIdx.x; c < DIM; c += 256) { const float v = x[c]; ss += v * v; }
    for (int off = 1; off < 32; off <<= 1) ss += __shfl_xor(ss, off, 32);
    __shared__ float red[8];
    if ((threadIdx.x & 31) == 0) red[threadIdx.x >> 5] = ss;
    __syncthreads();
    float tot = 0.0f;
#pragma unroll
    for (int i = 0; i < 8; ++i) tot += red[i];
    const float inv = rsqrtf(tot / (float)DIM + RMS_EPS);
    unsigned short* y = Yb + (size_t)row * DIM;
    for (int c = threadIdx.x; c < DIM; c += 256) y[c] = f2bf(x[c] * inv * W[c]);
}

__global__ __launch_bounds__(256) void rope_kernel(
    const float* __restrict__ X, const float* __restrict__ cosT,
    const float* __restrict__ sinT, const int* __restrict__ bi,
    const int* __restrict__ ti, unsigned short* __restrict__ Yb, int cols)
{
    const int n = blockIdx.x;
    const size_t crow = ((size_t)bi[n] * SEQ + ti[n]) * HDIM;
    const float* x = X + (size_t)n * cols;
    unsigned short* y = Yb + (size_t)n * cols;
    for (int c = threadIdx.x; c < cols; c += 256) {
        const int d = c & (HDIM - 1);
        const float cs = cosT[crow + d], sn = sinT[crow + d];
        const float rot = (d < HDIM / 2) ? -x[c + HDIM / 2] : x[c - HDIM / 2];
        y[c] = f2bf(x[c] * cs + rot * sn);
    }
}

__global__ __launch_bounds__(256) void f2bf_kernel(
    const float* __restrict__ X, unsigned short* __restrict__ Y, int nelem)
{
    const int i = blockIdx.x * 256 + threadIdx.x;
    if (i < nelem) Y[i] = f2bf(X[i]);
}

__global__ __launch_bounds__(256) void blend_scatter_kernel(
    const float* __restrict__ sel, const float* __restrict__ xf,
    const float* __restrict__ gate, const int* __restrict__ bi,
    const int* __restrict__ ti, float* __restrict__ out)
{
    const int n = blockIdx.x;
    const float g = gate[n];
    const size_t dst = ((size_t)bi[n] * SEQ + ti[n]) * DIM;
    for (int c = threadIdx.x; c < DIM; c += 256) {
        const size_t idx = (size_t)n * DIM + c;
        out[dst + c] = sel[idx] + g * (xf[idx] - sel[idx]);
    }
}

// =====================================================================
extern "C" void kernel_launch(void* const* d_in, const int* in_sizes, int n_in,
                              void* d_out, int out_size, void* d_ws, size_t ws_size,
                              hipStream_t stream) {
    const float* hidden = (const float*)d_in[0];
    const int*   bidx   = (const int*)d_in[1];
    const int*   tidx   = (const int*)d_in[2];
    const float* gate   = (const float*)d_in[3];
    const float* cosT   = (const float*)d_in[4];
    const float* sinT   = (const float*)d_in[5];
    const float* w_ln1  = (const float*)d_in[6];
    const float* w_ln2  = (const float*)d_in[7];
    const float* Wq     = (const float*)d_in[8];
    const float* bq     = (const float*)d_in[9];
    const float* Wk     = (const float*)d_in[10];
    const float* bk     = (const float*)d_in[11];
    const float* Wv     = (const float*)d_in[12];
    const float* bv     = (const float*)d_in[13];
    const float* Wo     = (const float*)d_in[14];
    const float* Wg     = (const float*)d_in[15];
    const float* Wu     = (const float*)d_in[16];
    const float* Wd     = (const float*)d_in[17];
    float* out = (float*)d_out;

    // workspace carve-out (~151 MB total)
    char* ws = (char*)d_ws;
    size_t off = 0;
    auto carve = [&](size_t bytes) -> char* {
        char* p = ws + off;
        off += (bytes + 255) & ~(size_t)255;
        return p;
    };
    float*          sel   = (float*)carve((size_t)NTOK * DIM * 4);
    float*          x     = (float*)carve((size_t)NTOK * DIM * 4);
    unsigned short* hbf   = (unsigned short*)carve((size_t)NTOK * DIM * 2);
    float*          qf32  = (float*)carve((size_t)NTOK * QCOLS * 4);
    float*          kf32  = (float*)carve((size_t)NTOK * KCOLS * 4);
    float*          vf32  = (float*)carve((size_t)NTOK * KCOLS * 4);
    unsigned short* qbf   = (unsigned short*)carve((size_t)NTOK * QCOLS * 2);
    unsigned short* kbf   = (unsigned short*)carve((size_t)NTOK * KCOLS * 2);
    unsigned short* vbf   = (unsigned short*)carve((size_t)NTOK * KCOLS * 2);
    float*          ctx   = (float*)carve((size_t)NTOK * QCOLS * 4);
    unsigned short* ctxbf = (unsigned short*)carve((size_t)NTOK * QCOLS * 2);
    unsigned short* gact  = (unsigned short*)carve((size_t)NTOK * ISZ * 2);
    unsigned short* act   = (unsigned short*)carve((size_t)NTOK * ISZ * 2);
    (void)ws_size; (void)in_sizes; (void)n_in; (void)out_size;

    // output starts as a copy of hidden_states; updated rows scattered later
    (void)hipMemcpyAsync(out, hidden, (size_t)BATCH * SEQ * DIM * sizeof(float),
                         hipMemcpyDeviceToDevice, stream);

    // 1) gather selected tokens
    gather_kernel<<<NTOK, 256, 0, stream>>>(hidden, bidx, tidx, sel, x);

    // 2) RMSNorm 1 -> bf16
    rmsnorm_kernel<<<NTOK, 256, 0, stream>>>(x, w_ln1, hbf);

    // 3) QKV projections (fp32 out, fused bias)
    gemm_bf16_kernel<0><<<dim3(QCOLS / 128, NTOK / 128), 256, 0, stream>>>(
        hbf, Wq, bq, nullptr, nullptr, qf32, nullptr, NTOK, QCOLS, DIM);
    gemm_bf16_kernel<0><<<dim3(KCOLS / 128, NTOK / 128), 256, 0, stream>>>(
        hbf, Wk, bk, nullptr, nullptr, kf32, nullptr, NTOK, KCOLS, DIM);
    gemm_bf16_kernel<0><<<dim3(KCOLS / 128, NTOK / 128), 256, 0, stream>>>(
        hbf, Wv, bv, nullptr, nullptr, vf32, nullptr, NTOK, KCOLS, DIM);

    // 4) RoPE (q, k) + bf16 convert; v straight convert
    rope_kernel<<<NTOK, 256, 0, stream>>>(qf32, cosT, sinT, bidx, tidx, qbf, QCOLS);
    rope_kernel<<<NTOK, 256, 0, stream>>>(kf32, cosT, sinT, bidx, tidx, kbf, KCOLS);
    f2bf_kernel<<<(NTOK * KCOLS) / 256, 256, 0, stream>>>(vf32, vbf, NTOK * KCOLS);

    // 5) attention (flash, GQA)
    attn_kernel<<<dim3(NTOK / 64, NH), 128, 0, stream>>>(qbf, kbf, vbf, ctx);

    // 6) output projection + residual (in place into x)
    f2bf_kernel<<<(NTOK * QCOLS) / 256, 256, 0, stream>>>(ctx, ctxbf, NTOK * QCOLS);
    gemm_bf16_kernel<0><<<dim3(DIM / 128, NTOK / 128), 256, 0, stream>>>(
        ctxbf, Wo, nullptr, x, nullptr, x, nullptr, NTOK, DIM, QCOLS);

    // 7) RMSNorm 2 -> bf16
    rmsnorm_kernel<<<NTOK, 256, 0, stream>>>(x, w_ln2, hbf);

    // 8) SwiGLU MLP: gact = silu(h2@Wg); act = gact * (h2@Wu); x += act@Wd
    gemm_bf16_kernel<2><<<dim3(ISZ / 128, NTOK / 128), 256, 0, stream>>>(
        hbf, Wg, nullptr, nullptr, nullptr, nullptr, gact, NTOK, ISZ, DIM);
    gemm_bf16_kernel<3><<<dim3(ISZ / 128, NTOK / 128), 256, 0, stream>>>(
        hbf, Wu, nullptr, nullptr, gact, nullptr, act, NTOK, ISZ, DIM);
    gemm_bf16_kernel<0><<<dim3(DIM / 128, NTOK / 128), 256, 0, stream>>>(
        act, Wd, nullptr, x, nullptr, x, nullptr, NTOK, DIM, ISZ);

    // 9) gated blend + scatter into output
    blend_scatter_kernel<<<NTOK, 256, 0, stream>>>(sel, x, gate, bidx, tidx, out);
}
